// Transformer_71485435675155
// MI455X (gfx1250) — compile-verified, matrix operand order checked by reading, not run on hardware
//
#include <hip/hip_runtime.h>
#include <math.h>

// ---------------------------------------------------------------------------
// Axial transformer block for MI455X (gfx1250, wave32, WMMA f16 16x16x32).
//
// B=8, L=512, E=256, NH=8, HD=32, H=W=128.
// mean(proj(x)) == proj(mean(x)) -> key_row/key_col projections shrink from
// 17.2 GF each to 0.13 GF each. Total ~55 GFLOP vs a ~420 MB / ~18us HBM
// floor, so: one f32 pass over inputs, all intermediates f16 (projected V =
// 67 MB, L2-resident), everything through v_wmma_f32_16x16x32_f16.
//
// Attention hot loop: V w-slices (8 KB) are staged global->LDS once per block
// (GLOBAL_LOAD_ASYNC_TO_LDS_B128, double-buffered, ASYNCcnt + barrier sync)
// and shared by all 4 waves -> 4x less L2/L0 read traffic in the hot loop.
// ---------------------------------------------------------------------------

typedef __attribute__((ext_vector_type(16))) _Float16 v16h;
typedef __attribute__((ext_vector_type(8)))  _Float16 v8h;
typedef __attribute__((ext_vector_type(8)))  float    v8f;
typedef int v4i __attribute__((vector_size(16)));

static constexpr int kB  = 8;
static constexpr int kL  = 512;
static constexpr int kE  = 256;
static constexpr int kNH = 8;
static constexpr int kHD = 32;
static constexpr int kH  = 128;
static constexpr int kW  = 128;

#define AS1 __attribute__((address_space(1)))
#define AS3 __attribute__((address_space(3)))

#if __has_builtin(__builtin_amdgcn_global_load_async_to_lds_b128) && \
    __has_builtin(__builtin_amdgcn_s_wait_asynccnt)
#define ASYNC_LDS_COPY 1
#else
#define ASYNC_LDS_COPY 0
#endif

__device__ __forceinline__ v8f wmma16(v16h a, v16h b, v8f c) {
  return __builtin_amdgcn_wmma_f32_16x16x32_f16(false, a, false, b,
                                                (short)0, c, false, false);
}

// f16 A/B fragment per ISA striping: lane<16 -> K 0..7 & 16..23,
// lane>=16 -> K 8..15 & 24..31 (kb0 = 0 or 8). Two b128 loads.
__device__ __forceinline__ v16h ldfrag(const _Float16* p, int kb0) {
  v8h lo = *(const v8h*)(p + kb0);
  v8h hi = *(const v8h*)(p + kb0 + 16);
  union { v8h h[2]; v16h v; } u;
  u.h[0] = lo; u.h[1] = hi;
  return u.v;
}

// Same fragment but source rows are f32 (converted on the fly).
__device__ __forceinline__ v16h ldfragA32(const float* p, int kb0) {
  const float4* q0 = (const float4*)(p + kb0);
  const float4* q1 = (const float4*)(p + kb0 + 16);
  float4 a0 = q0[0], a1 = q0[1], a2 = q1[0], a3 = q1[1];
  v16h r;
  r[0]  = (_Float16)a0.x; r[1]  = (_Float16)a0.y; r[2]  = (_Float16)a0.z; r[3]  = (_Float16)a0.w;
  r[4]  = (_Float16)a1.x; r[5]  = (_Float16)a1.y; r[6]  = (_Float16)a1.z; r[7]  = (_Float16)a1.w;
  r[8]  = (_Float16)a2.x; r[9]  = (_Float16)a2.y; r[10] = (_Float16)a2.z; r[11] = (_Float16)a2.w;
  r[12] = (_Float16)a3.x; r[13] = (_Float16)a3.y; r[14] = (_Float16)a3.z; r[15] = (_Float16)a3.w;
  return r;
}

// Cooperative 8 KB global->LDS stage: 128 threads x 4 x 16 B.
// Async path uses GLOBAL_LOAD_ASYNC_TO_LDS_B128 (ASYNCcnt, +4 per wave).
__device__ __forceinline__ void stage_slice(const _Float16* __restrict__ gsrc,
                                            _Float16* lbuf, int tid) {
#pragma unroll
  for (int i = 0; i < 4; ++i) {
    int off = i * 2048 + tid * 16;   // bytes, covers 8192
#if ASYNC_LDS_COPY
    __builtin_amdgcn_global_load_async_to_lds_b128(
        (AS1 v4i*)((const char*)gsrc + off),
        (AS3 v4i*)((char*)lbuf + off), 0, 0);
#else
    *(float4*)((char*)lbuf + off) = *(const float4*)((const char*)gsrc + off);
#endif
  }
}

__device__ __forceinline__ void wait_async_le4() {
#if ASYNC_LDS_COPY
  __builtin_amdgcn_s_wait_asynccnt(4);
#endif
}
__device__ __forceinline__ void wait_async_0() {
#if ASYNC_LDS_COPY
  __builtin_amdgcn_s_wait_asynccnt(0);
#endif
}

// ---------------------------------------------------------------------------
// Weight convert: in_proj_weight (5E x E) and out_proj_weight (E x E) -> f16,
// layout [n][k] with k contiguous (B-matrix fragments load straight out).
// ---------------------------------------------------------------------------
__global__ void wcvt_kernel(const float* __restrict__ ipw,
                            const float* __restrict__ opw,
                            _Float16* __restrict__ Wf) {
  int i = blockIdx.x * blockDim.x + threadIdx.x;   // 6*65536
  float v = (i < 5 * kE * kE) ? ipw[i] : opw[i - 5 * kE * kE];
  Wf[i] = (_Float16)v;
}

// mean over h: mrow[b][w][e] = mean_h key_row[b][h][w][e]
__global__ void mean_h_kernel(const float* __restrict__ keyr,
                              float* __restrict__ mrow) {
  int i = blockIdx.x * blockDim.x + threadIdx.x;   // B*W*E = 262144
  int e = i & (kE - 1);
  int w = (i >> 8) & (kW - 1);
  int b = i >> 15;
  const float* p = keyr + ((size_t)b * kH * kW + w) * kE + e;
  float s = 0.f;
#pragma unroll 4
  for (int h = 0; h < kH; ++h) s += p[(size_t)h * kW * kE];
  mrow[i] = s * (1.f / (float)kH);
}

// mean over w: mcol[b][h][e] = mean_w key_col[b][h][w][e]
__global__ void mean_w_kernel(const float* __restrict__ keyc,
                              float* __restrict__ mcol) {
  int i = blockIdx.x * blockDim.x + threadIdx.x;   // B*H*E
  int e = i & (kE - 1);
  int h = (i >> 8) & (kH - 1);
  int b = i >> 15;
  const float* p = keyc + ((size_t)(b * kH + h) * kW) * kE + e;
  float s = 0.f;
#pragma unroll 4
  for (int w = 0; w < kW; ++w) s += p[(size_t)w * kE];
  mcol[i] = s * (1.f / (float)kW);
}

// ---------------------------------------------------------------------------
// Generic projection GEMM: Y(f16, MxE) = ((A(f32, MxE) @ W^T) + bias) * alpha
// One wave per 16x64 tile; K=256 -> 8 WMMA k-steps, 4 n-tiles.
// ---------------------------------------------------------------------------
__global__ __launch_bounds__(128) void proj_gemm(const float* __restrict__ A,
                                                 const _Float16* __restrict__ W,
                                                 const float* __restrict__ bias,
                                                 float alpha,
                                                 _Float16* __restrict__ Y,
                                                 int M) {
  int wv   = blockIdx.x * (blockDim.x >> 5) + (threadIdx.x >> 5);
  int lane = threadIdx.x & 31;
  int lm   = lane & 15;
  int kb0  = (lane < 16) ? 0 : 8;
  int mb   = (lane < 16) ? 0 : 8;
  int mt   = wv >> 2;
  int nb   = wv & 3;
  (void)M;

  const float* arow = A + (size_t)(mt * 16 + lm) * kE;
  v8f acc[4] = {};
#pragma unroll
  for (int ks = 0; ks < 8; ++ks) {
    v16h af = ldfragA32(arow + ks * 32, kb0);
#pragma unroll
    for (int t = 0; t < 4; ++t) {
      int nc = nb * 64 + t * 16 + lm;
      v16h bf = ldfrag(W + (size_t)nc * kE + ks * 32, kb0);
      acc[t] = wmma16(af, bf, acc[t]);
    }
  }
#pragma unroll
  for (int t = 0; t < 4; ++t) {
    int nc = nb * 64 + t * 16 + lm;
    float bv = bias[nc];
#pragma unroll
    for (int r = 0; r < 8; ++r) {
      int mm = mt * 16 + mb + r;
      Y[(size_t)mm * kE + nc] = (_Float16)((acc[t][r] + bv) * alpha);
    }
  }
}

// ---------------------------------------------------------------------------
// Value projection with scattered store into VT[b][n][w][c][h] (h innermost)
// so attention B-fragments (K = h) are two b128 loads per lane.
// A rows m = ((b*H)+h)*W + w  (value is row-major (B,H,W,E)).
// ---------------------------------------------------------------------------
__global__ __launch_bounds__(128) void vproj_gemm(const float* __restrict__ A,
                                                  const _Float16* __restrict__ W,
                                                  const float* __restrict__ bias,
                                                  _Float16* __restrict__ VT) {
  int wv   = blockIdx.x * (blockDim.x >> 5) + (threadIdx.x >> 5);
  int lane = threadIdx.x & 31;
  int lm   = lane & 15;
  int kb0  = (lane < 16) ? 0 : 8;
  int mb   = (lane < 16) ? 0 : 8;
  int mt   = wv >> 2;
  int nb   = wv & 3;

  const float* arow = A + (size_t)(mt * 16 + lm) * kE;
  v8f acc[4] = {};
#pragma unroll
  for (int ks = 0; ks < 8; ++ks) {
    v16h af = ldfragA32(arow + ks * 32, kb0);
#pragma unroll
    for (int t = 0; t < 4; ++t) {
      int nc = nb * 64 + t * 16 + lm;
      v16h bf = ldfrag(W + (size_t)nc * kE + ks * 32, kb0);
      acc[t] = wmma16(af, bf, acc[t]);
    }
  }
#pragma unroll
  for (int t = 0; t < 4; ++t) {
    int nc = nb * 64 + t * 16 + lm;
    int nh = nc >> 5;          // head
    int c  = nc & 31;          // channel in head
    float bv = bias[nc];
#pragma unroll
    for (int r = 0; r < 8; ++r) {
      int mm = mt * 16 + mb + r;           // = (b*H + h)*W + w
      int b  = mm >> 14;                   // /(H*W)
      int rem = mm & (kH * kW - 1);
      int h  = rem >> 7;                   // /W
      int w  = rem & (kW - 1);
      VT[((((size_t)b * kNH + nh) * kW + w) * kHD + c) * kH + h] =
          (_Float16)(acc[t][r] + bv);
    }
  }
}

// ---------------------------------------------------------------------------
// Fused axial attention. One wave owns one (b, n, l-tile of 16 rows); the 4
// waves of a block share the same (b, n) head. Per l-tile:
//   scores_row/col (16x128) = q(16x32) @ k^T   -> 16 WMMAs, dumped to LDS
//   softmax per row (lanes 0-15: a_row, 16-31: a_col)
//   out(16x32) = sum_w a_row[:,w] * (a_col(16x128) @ V[:,w,:](128x32))
//     -> 8 WMMAs per w, V w-slice staged global->LDS once per block,
//        double-buffered (async-to-LDS + s_wait_asynccnt when available).
// ---------------------------------------------------------------------------
__global__ __launch_bounds__(128) void attn_kernel(const _Float16* __restrict__ qr,
                                                   const _Float16* __restrict__ qc,
                                                   const _Float16* __restrict__ kr,
                                                   const _Float16* __restrict__ kc,
                                                   const _Float16* __restrict__ VT,
                                                   _Float16* __restrict__ attnOut) {
  __shared__ float    sRow[4][kW * 16];              // scores / a_row (f32)
  __shared__ float    sCol[4][kH * 16];              // scores for a_col
  __shared__ _Float16 aCol[4][16 * kH];              // a_col, A-matrix layout
  __shared__ __align__(16) _Float16 vbuf[2][kHD * kH]; // V w-slice double buffer

  int tid  = threadIdx.x;
  int wid  = tid >> 5;
  int lane = tid & 31;
  int lm   = lane & 15;
  int kb0  = (lane < 16) ? 0 : 8;
  int mb   = (lane < 16) ? 0 : 8;

  int g  = blockIdx.x * 4 + wid;           // tile id in [0, B*NH*L/16)
  int lt = g & 31;
  int bn = g >> 5;                         // same for all 4 waves of a block
  int n  = bn & (kNH - 1);
  int b  = bn >> 3;
  int l0 = lt * 16;

  float*    sR = sRow[wid];
  float*    sC = sCol[wid];
  _Float16* aC = aCol[wid];

  // ----- scores (row & col) -----
  v16h aq = ldfrag(qr + ((size_t)(b * kL + l0 + lm) * kE) + n * kHD, kb0);
  v16h ac = ldfrag(qc + ((size_t)(b * kL + l0 + lm) * kE) + n * kHD, kb0);
#pragma unroll
  for (int t = 0; t < 8; ++t) {
    v16h bk = ldfrag(kr + ((size_t)(b * kW + t * 16 + lm) * kE) + n * kHD, kb0);
    v8f d = {};
    d = wmma16(aq, bk, d);
    float4* p = (float4*)&sR[(t * 16 + lm) * 16 + mb];
    p[0] = make_float4(d[0], d[1], d[2], d[3]);
    p[1] = make_float4(d[4], d[5], d[6], d[7]);

    v16h bc = ldfrag(kc + ((size_t)(b * kH + t * 16 + lm) * kE) + n * kHD, kb0);
    v8f d2 = {};
    d2 = wmma16(ac, bc, d2);
    float4* p2 = (float4*)&sC[(t * 16 + lm) * 16 + mb];
    p2[0] = make_float4(d2[0], d2[1], d2[2], d2[3]);
    p2[1] = make_float4(d2[4], d2[5], d2[6], d2[7]);
  }
  __syncthreads();

  // ----- softmax: lanes 0-15 own a_row rows, lanes 16-31 own a_col rows -----
  if (lane < 16) {
    int m = lane;
    float mx = -3.4e38f;
    for (int w = 0; w < kW; ++w) mx = fmaxf(mx, sR[w * 16 + m]);
    float s = 0.f;
    for (int w = 0; w < kW; ++w) {
      float e = __expf(sR[w * 16 + m] - mx);
      sR[w * 16 + m] = e;
      s += e;
    }
    float inv = 1.f / s;
    for (int w = 0; w < kW; ++w) sR[w * 16 + m] *= inv;
  } else {
    int m = lane - 16;
    float mx = -3.4e38f;
    for (int h = 0; h < kH; ++h) mx = fmaxf(mx, sC[h * 16 + m]);
    float s = 0.f;
    for (int h = 0; h < kH; ++h) {
      float e = __expf(sC[h * 16 + m] - mx);
      sC[h * 16 + m] = e;
      s += e;
    }
    float inv = 1.f / s;
    for (int h = 0; h < kH; ++h) aC[m * kH + h] = (_Float16)(sC[h * 16 + m] * inv);
  }
  __syncthreads();

  // ----- a_col A-fragments (K = 128 -> 4 k-steps), kept in VGPRs -----
  v16h af[4];
#pragma unroll
  for (int ks = 0; ks < 4; ++ks) af[ks] = ldfrag(&aC[lm * kH + ks * 32], kb0);

  const _Float16* vb = VT + (size_t)(b * kNH + n) * kW * kHD * kH;

  // Prime buffer 0 with w=0 slice (shared by all 4 waves).
  stage_slice(vb, vbuf[0], tid);

  v8f out0 = {}, out1 = {};
  for (int w = 0; w < kW; ++w) {
    if (w + 1 < kW) {
      stage_slice(vb + (size_t)(w + 1) * kHD * kH, vbuf[(w + 1) & 1], tid);
      wait_async_le4();      // own wave's stage(w) done, stage(w+1) in flight
    } else {
      wait_async_0();
    }
    __syncthreads();         // stage(w) visible from all waves

    const _Float16* vw = vbuf[w & 1];
    v8f t0 = {}, t1 = {};
#pragma unroll
    for (int ks = 0; ks < 4; ++ks) {
      v16h b0 = ldfrag(vw + (size_t)(lm) * kH + ks * 32, kb0);
      t0 = wmma16(af[ks], b0, t0);
      v16h b1 = ldfrag(vw + (size_t)(16 + lm) * kH + ks * 32, kb0);
      t1 = wmma16(af[ks], b1, t1);
    }
    const float4* arp = (const float4*)&sR[w * 16 + mb];
    float4 a0 = arp[0], a1 = arp[1];
    float ar[8] = {a0.x, a0.y, a0.z, a0.w, a1.x, a1.y, a1.z, a1.w};
#pragma unroll
    for (int i = 0; i < 8; ++i) {
      out0[i] += ar[i] * t0[i];
      out1[i] += ar[i] * t1[i];
    }
    __syncthreads();         // all waves done with buf[w&1] before overwrite
  }

  // ----- store attn_out[b][l][e] (f16) -----
#pragma unroll
  for (int r = 0; r < 8; ++r) {
    int mm = mb + r;
    size_t base = ((size_t)(b * kL) + (l0 + mm)) * kE + n * kHD;
    attnOut[base + lm]      = (_Float16)out0[r];
    attnOut[base + 16 + lm] = (_Float16)out1[r];
  }
}

// ---------------------------------------------------------------------------
// Output projection: d_out(L,B,E) f32 = attn(B,L,E) @ Wo^T + bias.
// Row remap: out row ro = l*B + b  <->  attn row = b*L + l.
// ---------------------------------------------------------------------------
__global__ __launch_bounds__(128) void out_proj_gemm(const _Float16* __restrict__ attn,
                                                     const _Float16* __restrict__ Wo,
                                                     const float* __restrict__ bias,
                                                     float* __restrict__ Y) {
  int wv   = blockIdx.x * (blockDim.x >> 5) + (threadIdx.x >> 5);
  int lane = threadIdx.x & 31;
  int lm   = lane & 15;
  int kb0  = (lane < 16) ? 0 : 8;
  int mb   = (lane < 16) ? 0 : 8;
  int mt   = wv >> 2;
  int nb   = wv & 3;

  int ro = mt * 16 + lm;                 // output row (l*B + b)
  int l  = ro >> 3;
  int b  = ro & (kB - 1);
  const _Float16* arow = attn + ((size_t)b * kL + l) * kE;

  v8f acc[4] = {};
#pragma unroll
  for (int ks = 0; ks < 8; ++ks) {
    v16h af = ldfrag(arow + ks * 32, kb0);
#pragma unroll
    for (int t = 0; t < 4; ++t) {
      int nc = nb * 64 + t * 16 + lm;
      v16h bf = ldfrag(Wo + (size_t)nc * kE + ks * 32, kb0);
      acc[t] = wmma16(af, bf, acc[t]);
    }
  }
#pragma unroll
  for (int t = 0; t < 4; ++t) {
    int nc = nb * 64 + t * 16 + lm;
    float bv = bias[nc];
#pragma unroll
    for (int r = 0; r < 8; ++r) {
      int rr = mt * 16 + mb + r;
      Y[(size_t)rr * kE + nc] = acc[t][r] + bv;
    }
  }
}

// ---------------------------------------------------------------------------
extern "C" void kernel_launch(void* const* d_in, const int* in_sizes, int n_in,
                              void* d_out, int out_size, void* d_ws, size_t ws_size,
                              hipStream_t stream) {
  const float* query_row = (const float*)d_in[0];
  const float* query_col = (const float*)d_in[1];
  const float* key_row   = (const float*)d_in[2];
  const float* key_col   = (const float*)d_in[3];
  const float* value     = (const float*)d_in[4];
  const float* ipw       = (const float*)d_in[5];
  const float* ipb       = (const float*)d_in[6];
  const float* opw       = (const float*)d_in[7];
  const float* opb       = (const float*)d_in[8];
  (void)in_sizes; (void)n_in; (void)out_size; (void)ws_size;

  char* ws = (char*)d_ws;
  size_t off = 0;
  auto carve = [&](size_t bytes) {
    size_t o = off;
    off = (off + bytes + 255) & ~(size_t)255;
    return o;
  };
  _Float16* Wf   = (_Float16*)(ws + carve((size_t)6 * kE * kE * 2));       // 768 KB
  float*    mrow = (float*)   (ws + carve((size_t)kB * kW * kE * 4));      // 1 MB
  float*    mcol = (float*)   (ws + carve((size_t)kB * kH * kE * 4));      // 1 MB
  _Float16* qr   = (_Float16*)(ws + carve((size_t)kB * kL * kE * 2));      // 2 MB
  _Float16* qc   = (_Float16*)(ws + carve((size_t)kB * kL * kE * 2));      // 2 MB
  _Float16* kr   = (_Float16*)(ws + carve((size_t)kB * kW * kE * 2));      // 512 KB
  _Float16* kc   = (_Float16*)(ws + carve((size_t)kB * kH * kE * 2));      // 512 KB
  _Float16* VT   = (_Float16*)(ws + carve((size_t)kB * kNH * kW * kHD * kH * 2)); // 67 MB
  _Float16* attn = (_Float16*)(ws + carve((size_t)kB * kL * kE * 2));      // 2 MB

  const float scale = 0.17677669529663687f;   // HD^-0.5 = 1/sqrt(32)

  wcvt_kernel<<<(6 * kE * kE) / 256, 256, 0, stream>>>(ipw, opw, Wf);
  mean_h_kernel<<<(kB * kW * kE) / 256, 256, 0, stream>>>(key_row, mrow);
  mean_w_kernel<<<(kB * kH * kE) / 256, 256, 0, stream>>>(key_col, mcol);

  // q_row / q_col: (x @ W^T + b) * scale, M = B*L = 4096 -> 1024 waves
  proj_gemm<<<256, 128, 0, stream>>>(query_row, Wf,               ipb,           scale, qr, kB * kL);
  proj_gemm<<<256, 128, 0, stream>>>(query_col, Wf + 1 * kE * kE, ipb + 1 * kE,  scale, qc, kB * kL);
  // k_row / k_col on the pre-averaged tensors, M = B*128 = 1024 -> 256 waves
  proj_gemm<<<64, 128, 0, stream>>>(mrow, Wf + 2 * kE * kE, ipb + 2 * kE, 1.f, kr, kB * kW);
  proj_gemm<<<64, 128, 0, stream>>>(mcol, Wf + 3 * kE * kE, ipb + 3 * kE, 1.f, kc, kB * kH);
  // value projection, M = B*H*W = 131072 -> 32768 waves
  vproj_gemm<<<8192, 128, 0, stream>>>(value, Wf + 4 * kE * kE, ipb + 4 * kE, VT);

  // fused axial attention: B*NH*(L/16) = 2048 wave-tiles, 4 per block
  attn_kernel<<<512, 128, 0, stream>>>(qr, qc, kr, kc, VT, attn);

  // output projection -> d_out (L, B, E) f32
  out_proj_gemm<<<256, 128, 0, stream>>>(attn, Wf + 5 * kE * kE, opb, (float*)d_out);
}
